// HumanNeRF_48249662603656
// MI455X (gfx1250) — matrix operand order repaired and checked by reading, not compiled, not run on hardware
//
#include <hip/hip_runtime.h>
#include <hip/hip_bf16.h>
#include <math.h>

// ---------------------------------------------------------------------------
// HumanNeRF forward, fused for MI455X (gfx1250, wave32, WMMA + TDM).
//  - Big MLP (~55 GFLOP dominant): v_wmma_f32_16x16x32_f16, activations in LDS,
//    per-block input tile DMA'd global->LDS via tensor_load_to_lds (TENSORcnt).
//  - 1-NN passes: v_wmma_f32_16x16x4_f32 on augmented vectors (exact f32 d^2
//    proxies), software-pipelined LDS B-fragment loads.
// ---------------------------------------------------------------------------

#define N_RAYS    1024
#define N_SAMPLES 64
#define P_TOTAL   (N_RAYS * N_SAMPLES)   // 65536
#define N_VERTS   6890
#define N_VPAD    6896                   // multiple of 16
#define IMG_W     512
#define FDIM      64

typedef __attribute__((ext_vector_type(16))) _Float16 v16h;
typedef __attribute__((ext_vector_type(8)))  _Float16 v8h;
typedef __attribute__((ext_vector_type(8)))  float    v8f;
typedef __attribute__((ext_vector_type(2)))  float    v2f;
typedef unsigned int u32x4 __attribute__((ext_vector_type(4)));
typedef int          i32x8 __attribute__((ext_vector_type(8)));
typedef int          i32x4 __attribute__((ext_vector_type(4)));

#if __has_builtin(__builtin_amdgcn_wmma_f32_16x16x4_f32)
#define KNN_USE_WMMA 1
#endif
#if __has_builtin(__builtin_amdgcn_tensor_load_to_lds) && \
    __has_builtin(__builtin_amdgcn_s_wait_tensorcnt)
#define HAVE_TDM 1
#endif

#ifdef HAVE_TDM
// One-row (1D) TDM descriptor: copy n contiguous f16 elements from global
// address g to LDS byte offset lds_off.  D# layout per CDNA5 ISA §8.3/8.4:
//   group0: count=1 | lds_addr | global_addr[56:0] | type=2
//   group1: data_size=1(2B); tensor_dim0=n; tensor_dim1=1; tile_dim0=n;
//           tile_dim1=1; tensor_dim0_stride=n
__device__ __forceinline__ void tdm_load_f16_1d(unsigned lds_off,
                                                const _Float16* g, unsigned n) {
  unsigned long long ga = (unsigned long long)(const void*)g;
  u32x4 g0;
  g0[0] = 1u;                                              // count=1, user D#
  g0[1] = lds_off;                                         // lds_addr (bytes)
  g0[2] = (unsigned)(ga & 0xffffffffu);                    // global_addr lo
  g0[3] = (unsigned)((ga >> 32) & 0x01ffffffu) | 0x80000000u; // addr hi | type=2
  i32x8 g1;
  g1[0] = 0x10000;                            // data_size=1 (2 bytes), mask=0
  g1[1] = (int)((n & 0xffffu) << 16);         // tensor_dim0[15:0]  (bits 63:48)
  g1[2] = (int)((n >> 16) & 0xffffu) | (1 << 16); // tensor_dim0[31:16] | tensor_dim1=1
  g1[3] = (int)((n & 0xffffu) << 16);         // tile_dim0 (bits 127:112), n<=65535
  g1[4] = 1;                                  // tile_dim1 = 1
  g1[5] = (int)n;                             // tensor_dim0_stride lo
  g1[6] = 0;
  g1[7] = 0;
  i32x4 z4 = {0, 0, 0, 0};
#if __clang_major__ >= 23
  i32x8 z8 = {0, 0, 0, 0, 0, 0, 0, 0};
  __builtin_amdgcn_tensor_load_to_lds(g0, g1, z4, z4, z8, 0);
#else
  __builtin_amdgcn_tensor_load_to_lds(g0, g1, z4, z4, 0);
#endif
}
#endif

// --- f16 WMMA fragment loaders (ISA 7.12.2 layouts) -------------------------
__device__ __forceinline__ v16h ldsA16(const _Float16* base, int pitch, int rb, int kt) {
  int lane = threadIdx.x & 31;
  int m  = lane & 15;
  int hi = lane >> 4;
  const _Float16* p = base + (size_t)(rb + m) * pitch + kt * 32 + hi * 8;
  v8h lo  = *(const v8h*)(p);
  v8h hi8 = *(const v8h*)(p + 16);
  v16h a;
#pragma unroll
  for (int i = 0; i < 8; ++i) { a[i] = lo[i]; a[i + 8] = hi8[i]; }
  return a;
}

__device__ __forceinline__ v16h ldsB16(const _Float16* wt, int pitch, int ntile, int kt) {
  int lane = threadIdx.x & 31;
  int n  = lane & 15;
  int hi = lane >> 4;
  const _Float16* p = wt + (size_t)(ntile * 16 + n) * pitch + kt * 32 + hi * 8;
  v8h x0 = *(const v8h*)(p);
  v8h x1 = *(const v8h*)(p + 16);
  v16h b;
#pragma unroll
  for (int i = 0; i < 8; ++i) { b[i] = x0[i]; b[i + 8] = x1[i]; }
  return b;
}

__device__ __forceinline__ v8f wmma_f16(v16h a, v16h b, v8f c) {
  return __builtin_amdgcn_wmma_f32_16x16x32_f16(false, a, false, b, (short)0, c, false, false);
}

// ---------------------------------------------------------------------------
// Kernel 0: sample points along rays.
// ---------------------------------------------------------------------------
__global__ void __launch_bounds__(256) k_points(
    const float* __restrict__ ro, const float* __restrict__ rd,
    const float* __restrict__ nearv, const float* __restrict__ farv,
    float* __restrict__ pts) {
  int g = blockIdx.x * 256 + threadIdx.x;
  if (g >= P_TOTAL) return;
  int ray = g >> 6, s = g & 63;
  float t = (float)s * (1.0f / (N_SAMPLES - 1));
  float z = nearv[ray] * (1.f - t) + farv[ray] * t;
#pragma unroll
  for (int c = 0; c < 3; ++c)
    pts[(size_t)g * 3 + c] = ro[ray * 3 + c] + rd[ray * 3 + c] * z;
}

// ---------------------------------------------------------------------------
// Kernel 1/2: brute-force 1-NN via f32 WMMA (16x16x4) on augmented vectors.
//   proxy[m][n] = |v_n|^2 - 2 q_m . v_n   (same argmin as d^2)
// B-fragment loads are software-pipelined one tile ahead so s_wait_dscnt
// overlaps the argmin VALU tail.
// ---------------------------------------------------------------------------
__global__ void __launch_bounds__(256) k_knn(
    const float* __restrict__ query,
    const float* __restrict__ refv,
    const float* __restrict__ addv,
    float* __restrict__ xout,
    float* __restrict__ d2out,
    float* __restrict__ visout,
    const int* __restrict__ vismask) {
  extern __shared__ float vaug[];   // [N_VPAD][4] = (x, y, z, |v|^2)
  int tid = threadIdx.x;
  for (int j = tid; j < N_VPAD; j += 256) {
    float x = 0.f, y = 0.f, z = 0.f, sq = 3.0e38f;
    if (j < N_VERTS) {
      x = refv[j * 3 + 0]; y = refv[j * 3 + 1]; z = refv[j * 3 + 2];
      sq = x * x + y * y + z * z;
    }
    vaug[j * 4 + 0] = x; vaug[j * 4 + 1] = y;
    vaug[j * 4 + 2] = z; vaug[j * 4 + 3] = sq;
  }
  __syncthreads();

  int wave = tid >> 5, lane = tid & 31;
  int qbase = blockIdx.x * 128 + wave * 16;
  int m = lane & 15, hi = lane >> 4;

#ifdef KNN_USE_WMMA
  const float* q = query + (size_t)(qbase + m) * 3;
  v2f afrag;
  if (hi == 0) { afrag.x = -2.f * q[0]; afrag.y = -2.f * q[1]; }
  else         { afrag.x = -2.f * q[2]; afrag.y = 1.f; }
#else
  float qx[8], qy[8], qz[8];
#pragma unroll
  for (int i = 0; i < 8; ++i) {
    const float* qq = query + (size_t)(qbase + hi * 8 + i) * 3;
    qx[i] = qq[0]; qy[i] = qq[1]; qz[i] = qq[2];
  }
#endif

  float bestv[8]; int besti[8];
#pragma unroll
  for (int i = 0; i < 8; ++i) { bestv[i] = 3.4e38f; besti[i] = 0; }

#ifdef KNN_USE_WMMA
  // pipelined: bfrag holds tile `tile`, prefetch tile+1 before consuming.
  v2f bfrag;
  { const float* vp = vaug + (size_t)m * 4 + hi * 2; bfrag.x = vp[0]; bfrag.y = vp[1]; }
  for (int tile = 0; tile < N_VPAD / 16; ++tile) {
    int vidx = tile * 16 + m;
    v8f c = {};
    c = __builtin_amdgcn_wmma_f32_16x16x4_f32(false, afrag, false, bfrag,
                                              (short)0, c, false, false);
    if (tile + 1 < N_VPAD / 16) {
      const float* vp = vaug + (size_t)(vidx + 16) * 4 + hi * 2;
      bfrag.x = vp[0]; bfrag.y = vp[1];
    }
#pragma unroll
    for (int i = 0; i < 8; ++i) {
      float d = c[i];
      if (d < bestv[i]) { bestv[i] = d; besti[i] = vidx; }
    }
  }
#else
  for (int tile = 0; tile < N_VPAD / 16; ++tile) {
    int vidx = tile * 16 + m;
    const float* vp = vaug + (size_t)vidx * 4;
    float vx = vp[0], vy = vp[1], vz = vp[2], vs = vp[3];
#pragma unroll
    for (int i = 0; i < 8; ++i) {
      float d = vs - 2.f * (qx[i] * vx + qy[i] * vy + qz[i] * vz);
      if (d < bestv[i]) { bestv[i] = d; besti[i] = vidx; }
    }
  }
#endif

  // Min-index reduce across the 16 lanes holding the same row.
#pragma unroll
  for (int i = 0; i < 8; ++i) {
#pragma unroll
    for (int off = 8; off >= 1; off >>= 1) {
      float ov = __shfl_xor(bestv[i], off, 32);
      int   oi = __shfl_xor(besti[i], off, 32);
      if (ov < bestv[i] || (ov == bestv[i] && oi < besti[i])) {
        bestv[i] = ov; besti[i] = oi;
      }
    }
  }
  if (m == 0) {   // lanes 0 (rows 0-7) and 16 (rows 8-15)
#pragma unroll
    for (int i = 0; i < 8; ++i) {
      int r = hi * 8 + i;
      int g = qbase + r;
      int bi = besti[i];
      const float* qq = query + (size_t)g * 3;
      float px = qq[0], py = qq[1], pz = qq[2];
      xout[(size_t)g * 3 + 0] = px - refv[bi * 3 + 0] + addv[bi * 3 + 0];
      xout[(size_t)g * 3 + 1] = py - refv[bi * 3 + 1] + addv[bi * 3 + 1];
      xout[(size_t)g * 3 + 2] = pz - refv[bi * 3 + 2] + addv[bi * 3 + 2];
      if (d2out)  d2out[g]  = bestv[i] + px * px + py * py + pz * pz;
      if (visout) visout[g] = (float)vismask[bi];
    }
  }
}

// ---------------------------------------------------------------------------
// Kernel 3: projection + bilinear sampling + pixel MLP 96->64->64->32 (WMMA),
// emits the 96-wide big-MLP feature as f16.  Block = 128 points, 8 waves.
// ---------------------------------------------------------------------------
#define U_PITCH 104
#define H_PITCH 72

__global__ void __launch_bounds__(256) k_pixmlp(
    const float* __restrict__ xobs, const float* __restrict__ xskel,
    const float* __restrict__ visbuf,
    const float* __restrict__ img, const float* __restrict__ fmap,
    const float* __restrict__ E, const float* __restrict__ K,
    const float* __restrict__ pw1, const float* __restrict__ pb1,
    const float* __restrict__ pw2, const float* __restrict__ pb2,
    const float* __restrict__ pw3, const float* __restrict__ pb3,
    _Float16* __restrict__ xfeat) {
  extern __shared__ _Float16 sm3[];
  _Float16* uM  = sm3;                          // [128][U_PITCH]
  _Float16* w1t = uM  + 128 * U_PITCH;          // [64][U_PITCH]
  _Float16* w2t = w1t + 64  * U_PITCH;          // [64][H_PITCH]
  _Float16* w3t = w2t + 64  * H_PITCH;          // [32][H_PITCH]
  _Float16* h1  = w3t + 32  * H_PITCH;          // [128][H_PITCH]
  _Float16* h2  = h1  + 128 * H_PITCH;          // [128][H_PITCH]

  int tid = threadIdx.x;
  size_t gb = (size_t)blockIdx.x * 128;

  for (int e = tid; e < 64 * 96; e += 256) {
    int n = e / 96, k = e % 96;
    w1t[n * U_PITCH + k] = (_Float16)((k < 91) ? pw1[k * 64 + n] : 0.f);
  }
  for (int e = tid; e < 64 * 64; e += 256) {
    int n = e >> 6, k = e & 63;
    w2t[n * H_PITCH + k] = (_Float16)pw2[k * 64 + n];
  }
  for (int e = tid; e < 32 * 64; e += 256) {
    int n = e >> 6, k = e & 63;
    w3t[n * H_PITCH + k] = (_Float16)pw3[k * 32 + n];
  }

  if (tid < 128) {
    int g = (int)gb + tid;
    float X0 = xobs[(size_t)g * 3 + 0], X1 = xobs[(size_t)g * 3 + 1],
          X2 = xobs[(size_t)g * 3 + 2];
    float cx = E[0] * X0 + E[1] * X1 + E[2]  * X2 + E[3];
    float cy = E[4] * X0 + E[5] * X1 + E[6]  * X2 + E[7];
    float cz = E[8] * X0 + E[9] * X1 + E[10] * X2 + E[11];
    float uh = K[0] * cx + K[1] * cy + K[2] * cz;
    float vh = K[3] * cx + K[4] * cy + K[5] * cz;
    float wh = K[6] * cx + K[7] * cy + K[8] * cz;
    float uu = uh / wh, vv = vh / wh;
    float fx = uu * ((float)(IMG_W - 1) / (float)IMG_W);
    float fy = vv * ((float)(IMG_W - 1) / (float)IMG_W);
    float x0f = floorf(fx), y0f = floorf(fy);
    float wx1 = fx - x0f, wy1 = fy - y0f;
    int x0 = (int)x0f, y0 = (int)y0f;
    int xs[2] = {x0, x0 + 1}, ys[2] = {y0, y0 + 1};
    float wx[2] = {1.f - wx1, wx1}, wy[2] = {1.f - wy1, wy1};
    float cw[4]; int co[4];
#pragma unroll
    for (int j = 0; j < 4; ++j) {
      int xi = xs[j & 1], yi = ys[j >> 1];
      bool ok = (xi >= 0) && (xi < IMG_W) && (yi >= 0) && (yi < IMG_W);
      int xc = min(max(xi, 0), IMG_W - 1), yc = min(max(yi, 0), IMG_W - 1);
      cw[j] = ok ? wx[j & 1] * wy[j >> 1] : 0.f;
      co[j] = yc * IMG_W + xc;
    }
    float vis = visbuf[g];
    _Float16* ur = uM + (size_t)tid * U_PITCH;
    float rgb[3];
#pragma unroll
    for (int c = 0; c < 3; ++c) {
      const float* ch = img + (size_t)c * IMG_W * IMG_W;
      float v = ch[co[0]] * cw[0] + ch[co[1]] * cw[1] +
                ch[co[2]] * cw[2] + ch[co[3]] * cw[3];
      rgb[c] = v;
      ur[c] = (_Float16)(v * vis);
    }
#pragma unroll
    for (int f = 0; f < 4; ++f) {
      float fq = (float)(1 << f);
#pragma unroll
      for (int c = 0; c < 3; ++c) {
        float a = fq * rgb[c];
        ur[3 + f * 3 + c]  = (_Float16)(sinf(a) * vis);
        ur[15 + f * 3 + c] = (_Float16)(cosf(a) * vis);
      }
    }
    for (int c = 0; c < FDIM; ++c) {
      const float* ch = fmap + (size_t)c * IMG_W * IMG_W;
      float v = ch[co[0]] * cw[0] + ch[co[1]] * cw[1] +
                ch[co[2]] * cw[2] + ch[co[3]] * cw[3];
      ur[27 + c] = (_Float16)(v * vis);
    }
#pragma unroll
    for (int k = 91; k < 96; ++k) ur[k] = (_Float16)0.f;

    _Float16* xf = xfeat + (size_t)g * 96;
    float S[3] = {xskel[(size_t)g * 3], xskel[(size_t)g * 3 + 1],
                  xskel[(size_t)g * 3 + 2]};
#pragma unroll
    for (int c = 0; c < 3; ++c) xf[c] = (_Float16)S[c];
#pragma unroll
    for (int f = 0; f < 10; ++f) {
      float fq = (float)(1 << f);
#pragma unroll
      for (int c = 0; c < 3; ++c) {
        float a = fq * S[c];
        xf[3  + f * 3 + c] = (_Float16)sinf(a);
        xf[33 + f * 3 + c] = (_Float16)cosf(a);
      }
    }
    xf[95] = (_Float16)0.f;
  }
  __syncthreads();

  int wave = tid >> 5, lane = tid & 31;
  int rb = wave * 16, n16 = lane & 15, hi = lane >> 4;

  for (int nt = 0; nt < 4; ++nt) {               // 96 -> 64
    v8f acc = {};
#pragma unroll
    for (int kt = 0; kt < 3; ++kt)
      acc = wmma_f16(ldsA16(uM, U_PITCH, rb, kt), ldsB16(w1t, U_PITCH, nt, kt), acc);
    int n = nt * 16 + n16;
    float bias = pb1[n];
#pragma unroll
    for (int i = 0; i < 8; ++i)
      h1[(size_t)(rb + hi * 8 + i) * H_PITCH + n] =
          (_Float16)fmaxf(acc[i] + bias, 0.f);
  }
  for (int nt = 0; nt < 4; ++nt) {               // 64 -> 64
    v8f acc = {};
#pragma unroll
    for (int kt = 0; kt < 2; ++kt)
      acc = wmma_f16(ldsA16(h1, H_PITCH, rb, kt), ldsB16(w2t, H_PITCH, nt, kt), acc);
    int n = nt * 16 + n16;
    float bias = pb2[n];
#pragma unroll
    for (int i = 0; i < 8; ++i)
      h2[(size_t)(rb + hi * 8 + i) * H_PITCH + n] =
          (_Float16)fmaxf(acc[i] + bias, 0.f);
  }
  for (int nt = 0; nt < 2; ++nt) {               // 64 -> 32 (no relu)
    v8f acc = {};
#pragma unroll
    for (int kt = 0; kt < 2; ++kt)
      acc = wmma_f16(ldsA16(h2, H_PITCH, rb, kt), ldsB16(w3t, H_PITCH, nt, kt), acc);
    int n = nt * 16 + n16;
    float bias = pb3[n];
#pragma unroll
    for (int i = 0; i < 8; ++i) {
      int r = rb + hi * 8 + i;
      xfeat[(gb + r) * 96 + 63 + n] = (_Float16)(acc[i] + bias);
    }
  }
}

// ---------------------------------------------------------------------------
// Kernel 4: fused big MLP 96 -> 256 -> (6x)256 -> 4.  Block = 128 points.
// Input tile arrives via tensor_load_to_lds into packed xin[128][96] (LDS
// offset 0); w_in staging overlaps the DMA; completion via s_wait_tensorcnt.
// LDS: 24KB xin + 2x66KB activations + 132KB weights = 288KB <= 320KB/WGP.
// ---------------------------------------------------------------------------
#define A_PITCH 264
#define XIN_PITCH 96

__global__ void __launch_bounds__(256) k_bigmlp(
    const _Float16* __restrict__ xfeat,
    const float* __restrict__ w_in,  const float* __restrict__ b_in,
    const float* __restrict__ w_hid, const float* __restrict__ b_hid,
    const float* __restrict__ w_out, const float* __restrict__ b_out,
    float* __restrict__ raw) {
  extern __shared__ _Float16 sm4[];
  _Float16* xin  = sm4;                       // [128][96]   packed, TDM dest @ 0
  _Float16* actA = xin  + 128 * XIN_PITCH;    // [128][A_PITCH]
  _Float16* actB = actA + 128 * A_PITCH;      // [128][A_PITCH]
  _Float16* wt   = actB + 128 * A_PITCH;      // [256][A_PITCH]

  int tid = threadIdx.x;
  size_t gb = (size_t)blockIdx.x * 128;

#ifdef HAVE_TDM
  if (tid < 32)   // one wave issues the tensor DMA (EXEC-independent issue)
    tdm_load_f16_1d(0u, xfeat + gb * XIN_PITCH, 128 * XIN_PITCH);
#endif
  // overlap: stage w_in^T (f32 -> f16, zero-pad k=95) while the DMA runs
  for (int e = tid; e < 256 * 96; e += 256) {
    int n = e / 96, k = e % 96;
    wt[(size_t)n * A_PITCH + k] = (_Float16)((k < 95) ? w_in[k * 256 + n] : 0.f);
  }
  __builtin_prefetch(w_hid + tid * 64, 0, 1);
#ifdef HAVE_TDM
  if (tid < 32) __builtin_amdgcn_s_wait_tensorcnt(0);
#else
  for (int e = tid; e < 128 * XIN_PITCH; e += 256)
    xin[e] = xfeat[gb * XIN_PITCH + e];
#endif
  __syncthreads();

  int wave = tid >> 5, lane = tid & 31;
  int rb = wave * 16, n16 = lane & 15, hi = lane >> 4;

  // input layer: K=96 (3 ktiles), N=256 (16 ntiles), A from packed xin
  for (int nt = 0; nt < 16; ++nt) {
    v8f acc = {};
#pragma unroll
    for (int kt = 0; kt < 3; ++kt)
      acc = wmma_f16(ldsA16(xin, XIN_PITCH, rb, kt), ldsB16(wt, A_PITCH, nt, kt), acc);
    int n = nt * 16 + n16;
    float bias = b_in[n];
#pragma unroll
    for (int i = 0; i < 8; ++i)
      actA[(size_t)(rb + hi * 8 + i) * A_PITCH + n] =
          (_Float16)fmaxf(acc[i] + bias, 0.f);
  }
  __syncthreads();

  _Float16* A = actA; _Float16* B = actB;
  for (int layer = 0; layer < 6; ++layer) {
    const float* W = w_hid + (size_t)layer * 256 * 256;
    for (int e = tid; e < 256 * 256; e += 256) {
      int n = e >> 8, k = e & 255;
      wt[(size_t)n * A_PITCH + k] = (_Float16)W[k * 256 + n];
    }
    if (layer + 1 < 6)
      __builtin_prefetch(w_hid + (size_t)(layer + 1) * 65536 + tid * 64, 0, 1);
    __syncthreads();
    for (int nt = 0; nt < 16; ++nt) {
      v8f acc = {};
#pragma unroll
      for (int kt = 0; kt < 8; ++kt)
        acc = wmma_f16(ldsA16(A, A_PITCH, rb, kt), ldsB16(wt, A_PITCH, nt, kt), acc);
      int n = nt * 16 + n16;
      float bias = b_hid[layer * 256 + n];
#pragma unroll
      for (int i = 0; i < 8; ++i)
        B[(size_t)(rb + hi * 8 + i) * A_PITCH + n] =
            (_Float16)fmaxf(acc[i] + bias, 0.f);
    }
    __syncthreads();
    _Float16* t = A; A = B; B = t;
  }

  // output layer 256 -> 4 (one N-tile, cols 4..15 zero)
  for (int e = tid; e < 16 * 256; e += 256) {
    int n = e >> 8, k = e & 255;
    wt[(size_t)n * A_PITCH + k] = (_Float16)((n < 4) ? w_out[k * 4 + n] : 0.f);
  }
  __syncthreads();
  v8f acc = {};
#pragma unroll
  for (int kt = 0; kt < 8; ++kt)
    acc = wmma_f16(ldsA16(A, A_PITCH, rb, kt), ldsB16(wt, A_PITCH, 0, kt), acc);
  if (n16 < 4) {
    float bias = b_out[n16];
#pragma unroll
    for (int i = 0; i < 8; ++i) {
      int r = rb + hi * 8 + i;
      raw[(gb + r) * 4 + n16] = acc[i] + bias;
    }
  }
}

// ---------------------------------------------------------------------------
// Kernel 5: per-ray volume rendering.
// ---------------------------------------------------------------------------
__global__ void __launch_bounds__(256) k_render(
    const float* __restrict__ raw, const float* __restrict__ d2buf,
    const float* __restrict__ nearv, const float* __restrict__ farv,
    const float* __restrict__ rd, const float* __restrict__ bg,
    float* __restrict__ out) {
  int ray = blockIdx.x * 256 + threadIdx.x;
  if (ray >= N_RAYS) return;
  float nr = nearv[ray], fr = farv[ray];
  float d0 = rd[ray * 3], d1 = rd[ray * 3 + 1], d2 = rd[ray * 3 + 2];
  float dn = sqrtf(d0 * d0 + d1 * d1 + d2 * d2);
  float trans = 1.f, accw = 0.f, cr = 0.f, cg = 0.f, cb = 0.f;
  for (int s = 0; s < N_SAMPLES; ++s) {
    float t  = (float)s * (1.f / (N_SAMPLES - 1));
    float z  = nr * (1.f - t) + fr * t;
    float dist;
    if (s == N_SAMPLES - 1) dist = 1e10f;
    else {
      float t1 = (float)(s + 1) * (1.f / (N_SAMPLES - 1));
      dist = (nr * (1.f - t1) + fr * t1) - z;
    }
    dist *= dn;
    size_t g = (size_t)ray * N_SAMPLES + s;
    const float* rw = raw + g * 4;
    float mask = (d2buf[g] < 0.0025f) ? 1.f : 0.f;
    float sp = rw[3] - 1.f;
    sp = (sp > 20.f) ? sp : log1pf(expf(sp));
    float alpha = (1.f - expf(-sp * dist)) * mask;
    float w = alpha * trans;
    float r  = 1.f / (1.f + expf(-rw[0])) * 1.002f - 0.001f;
    float gg = 1.f / (1.f + expf(-rw[1])) * 1.002f - 0.001f;
    float b  = 1.f / (1.f + expf(-rw[2])) * 1.002f - 0.001f;
    cr += w * r; cg += w * gg; cb += w * b;
    accw += w;
    trans *= (1.f - alpha + 1e-10f);
  }
  float rest = 1.f - accw;
  out[ray * 3 + 0] = cr + rest * bg[0] * (1.f / 255.f);
  out[ray * 3 + 1] = cg + rest * bg[1] * (1.f / 255.f);
  out[ray * 3 + 2] = cb + rest * bg[2] * (1.f / 255.f);
}

// ---------------------------------------------------------------------------
extern "C" void kernel_launch(void* const* d_in, const int* in_sizes, int n_in,
                              void* d_out, int out_size, void* d_ws, size_t ws_size,
                              hipStream_t stream) {
  const float* rays_o   = (const float*)d_in[0];
  const float* rays_d   = (const float*)d_in[1];
  const float* nearv    = (const float*)d_in[2];
  const float* farv     = (const float*)d_in[3];
  const float* verts_t  = (const float*)d_in[4];
  const float* verts_c  = (const float*)d_in[5];
  const float* verts_o  = (const float*)d_in[6];
  const float* img      = (const float*)d_in[7];
  const float* fmap     = (const float*)d_in[8];
  const float* extr     = (const float*)d_in[9];
  const float* intr     = (const float*)d_in[10];
  const float* bgcolor  = (const float*)d_in[11];
  const float* pw1      = (const float*)d_in[12];
  const float* pb1      = (const float*)d_in[13];
  const float* pw2      = (const float*)d_in[14];
  const float* pb2      = (const float*)d_in[15];
  const float* pw3      = (const float*)d_in[16];
  const float* pb3      = (const float*)d_in[17];
  const float* w_in     = (const float*)d_in[18];
  const float* b_in     = (const float*)d_in[19];
  const float* w_hid    = (const float*)d_in[20];
  const float* b_hid    = (const float*)d_in[21];
  const float* w_out    = (const float*)d_in[22];
  const float* b_out    = (const float*)d_in[23];
  const int*   vis_mask = (const int*)d_in[24];
  float* out = (float*)d_out;

  char* ws = (char*)d_ws;
  float* pts    = (float*)ws;                 ws += (size_t)P_TOTAL * 3 * sizeof(float);
  float* xskel  = (float*)ws;                 ws += (size_t)P_TOTAL * 3 * sizeof(float);
  float* xobs   = (float*)ws;                 ws += (size_t)P_TOTAL * 3 * sizeof(float);
  float* d2buf  = (float*)ws;                 ws += (size_t)P_TOTAL * sizeof(float);
  float* visbuf = (float*)ws;                 ws += (size_t)P_TOTAL * sizeof(float);
  _Float16* xfeat = (_Float16*)ws;            ws += (size_t)P_TOTAL * 96 * sizeof(_Float16);
  float* rawbuf = (float*)ws;                 ws += (size_t)P_TOTAL * 4 * sizeof(float);

  k_points<<<P_TOTAL / 256, 256, 0, stream>>>(rays_o, rays_d, nearv, farv, pts);

  size_t knn_lds = (size_t)N_VPAD * 4 * sizeof(float);   // ~110 KB
  k_knn<<<P_TOTAL / 128, 256, knn_lds, stream>>>(
      pts, verts_t, verts_c, xskel, d2buf, visbuf, vis_mask);
  k_knn<<<P_TOTAL / 128, 256, knn_lds, stream>>>(
      xskel, verts_c, verts_o, xobs, nullptr, nullptr, nullptr);

  size_t k3_lds = (size_t)(128 * U_PITCH + 64 * U_PITCH + 64 * H_PITCH +
                           32 * H_PITCH + 128 * H_PITCH + 128 * H_PITCH) *
                  sizeof(_Float16);                       // ~90 KB
  k_pixmlp<<<P_TOTAL / 128, 256, k3_lds, stream>>>(
      xobs, xskel, visbuf, img, fmap, extr, intr,
      pw1, pb1, pw2, pb2, pw3, pb3, xfeat);

  size_t k4_lds = (size_t)(128 * XIN_PITCH + 2 * 128 * A_PITCH + 256 * A_PITCH) *
                  sizeof(_Float16);                       // ~288 KB
  k_bigmlp<<<P_TOTAL / 128, 256, k4_lds, stream>>>(
      xfeat, w_in, b_in, w_hid, b_hid, w_out, b_out, rawbuf);

  k_render<<<(N_RAYS + 255) / 256, 256, 0, stream>>>(
      rawbuf, d2buf, nearv, farv, rays_d, bgcolor, out);
}
